// TTQConv2d_75153337745917
// MI455X (gfx1250) — matrix operand order repaired and checked by reading, not compiled, not run on hardware
//
#include <hip/hip_runtime.h>
#include <hip/hip_bf16.h>
#include <stdint.h>

typedef __attribute__((ext_vector_type(16))) _Float16 v16h;
typedef __attribute__((ext_vector_type(8)))  _Float16 v8h;
typedef __attribute__((ext_vector_type(4)))  _Float16 v4h;
typedef __attribute__((ext_vector_type(8)))  float    v8f;
typedef __attribute__((ext_vector_type(4)))  unsigned int v4u;
typedef __attribute__((ext_vector_type(8)))  int      v8i;
typedef __attribute__((ext_vector_type(4)))  int      v4i;

#define C_IN    256
#define K_OUT   256
#define HW_IMG  3136      // 56*56
#define W_IMG   56
#define KGEMM   2304      // 256*3*3
#define MTILE   256       // all output channels -> x read exactly once
#define NTILE   64
#define KSTEP   64
#define LSTRIDE 72        // halfs; 144B row stride -> 16B aligned, conflict-free
#define NWEIGHT (K_OUT * C_IN * 9)   // 589824
#define ATILE_HALFS (MTILE * LSTRIDE)          // 18432 halfs = 36864 B

// ---------- pass 1: max(|w|) two-stage reduction ----------
__global__ __launch_bounds__(256) void ttq_absmax_stage1(
    const float* __restrict__ w, int n, float* __restrict__ partials) {
  __shared__ float sm[256];
  int tid = threadIdx.x;
  float m = 0.0f;
  for (int i = blockIdx.x * 256 + tid; i < n; i += gridDim.x * 256)
    m = fmaxf(m, fabsf(w[i]));
  sm[tid] = m;
  __syncthreads();
  for (int s = 128; s > 0; s >>= 1) {
    if (tid < s) sm[tid] = fmaxf(sm[tid], sm[tid + s]);
    __syncthreads();
  }
  if (tid == 0) partials[blockIdx.x] = sm[0];
}

__global__ __launch_bounds__(256) void ttq_absmax_stage2(float* __restrict__ wsf) {
  __shared__ float sm[256];
  int tid = threadIdx.x;
  sm[tid] = fmaxf(wsf[tid], wsf[tid + 256]);
  __syncthreads();
  for (int s = 128; s > 0; s >>= 1) {
    if (tid < s) sm[tid] = fmaxf(sm[tid], sm[tid + s]);
    __syncthreads();
  }
  if (tid == 0) wsf[512] = sm[0];
}

// ---------- pass 2: quantize weights once to f16 ternary ----------
__global__ __launch_bounds__(256) void ttq_quantize(
    const float* __restrict__ wgt, const float* __restrict__ maxptr,
    const float* __restrict__ Wp, const float* __restrict__ Wn,
    _Float16* __restrict__ wq) {
  const float delta = 0.05f * maxptr[0];
  const float wp = Wp[0];
  const float wn = Wn[0];
  int idx = (blockIdx.x * 256 + threadIdx.x) * 4;   // 576 blocks cover 589824
  float4 w = *(const float4*)&wgt[idx];
  v4h q;
  q[0] = (_Float16)((w.x > delta) ? wp : ((w.x < -delta) ? -wn : 0.0f));
  q[1] = (_Float16)((w.y > delta) ? wp : ((w.y < -delta) ? -wn : 0.0f));
  q[2] = (_Float16)((w.z > delta) ? wp : ((w.z < -delta) ? -wn : 0.0f));
  q[3] = (_Float16)((w.w > delta) ? wp : ((w.w < -delta) ? -wn : 0.0f));
  *(v4h*)&wq[idx] = q;
}

// ---------- pass 3: implicit-GEMM WMMA conv ----------
// PRE=true : A-tile DMA'd from pre-quantized f16 wq via TDM, double-buffered
// PRE=false: A-tile quantized on the fly from f32 weights (ws too small)
template <bool PRE>
__global__ __launch_bounds__(256) void ttq_conv_wmma(
    const float* __restrict__ x, const float* __restrict__ wgt,
    const _Float16* __restrict__ wq, const float* __restrict__ bias,
    const float* __restrict__ Wp, const float* __restrict__ Wn,
    const float* __restrict__ maxptr, float* __restrict__ out) {
  __shared__ alignas(16) _Float16 Alds[2][ATILE_HALFS]; // [buf][chan*LSTRIDE+k]
  __shared__ alignas(16) _Float16 Blds[NTILE * LSTRIDE];

  const int tid     = threadIdx.x;
  const int pixBase = blockIdx.x * NTILE;   // 1568 blocks

  const int wv   = tid >> 5;     // wave 0..7
  const int lane = tid & 31;
  const int g    = lane >> 4;    // lane group
  const int ln   = lane & 15;
  const int wnt  = wv & 1;       // N-position: pixels [wnt*32, wnt*32+32)
  const int wm   = wv >> 1;      // M-position: rows   [wm*64,  wm*64+64)

  // ---- stage-invariant thread coordinates ----
  const int sub  = tid & 63;     // B pixel / fallback-A k-column (fixed)
  const int quad = tid >> 6;     // 0..3
  const int pixS = pixBase + sub;
  const int nbS  = pixS / HW_IMG;
  const int hwS  = pixS - nbS * HW_IMG;
  const int hS   = hwS / W_IMG;
  const int wS   = hwS - hS * W_IMG;
  const long xBase = (long)nbS * C_IN * HW_IMG + hS * W_IMG + wS;

  float delta = 0.0f, wp = 0.0f, wn = 0.0f;
  if constexpr (!PRE) {
    delta = 0.05f * maxptr[0];
    wp = Wp[0];
    wn = Wn[0];
  }

  // LDS byte offset of A buffers (generic shared addr low 32 bits = LDS offset)
  const unsigned ldsABase = (unsigned)(uintptr_t)&Alds[0][0];

  // TDM: load 256x64 f16 tile of wq (row stride 2304) into LDS with 16B row pad
  auto tdmIssueA = [&](int kkT, int buf) {
    if constexpr (PRE) {
      if (wv == 0) {
        unsigned long long ga =
            (unsigned long long)(uintptr_t)wq + (unsigned long long)kkT * 2ull;
        v4u g0;
        g0[0] = 1u;                                  // count=1, user mode
        g0[1] = ldsABase + (unsigned)buf * (ATILE_HALFS * 2);
        g0[2] = (unsigned)(ga & 0xffffffffull);
        g0[3] = (unsigned)((ga >> 32) & 0x1ffffffull) | (2u << 30); // type=2
        v8i g1;
        g1[0] = (int)((1u << 16)      // data_size: 2 bytes
                    | (1u << 20)      // pad_enable
                    | (4u << 22)      // pad_interval: 32 DWORDs (128 B)
                    | (3u << 25));    // pad_amount: 4 DWORDs (16 B)
        g1[1] = (int)((unsigned)KGEMM << 16);  // tensor_dim0[15:0]
        g1[2] = (int)((unsigned)MTILE << 16);  // tensor_dim0 hi=0, tensor_dim1 lo
        g1[3] = (int)((unsigned)KSTEP << 16);  // tensor_dim1 hi=0, tile_dim0=64
        g1[4] = MTILE;                         // tile_dim1=256, tile_dim2=0
        g1[5] = KGEMM;                         // tensor_dim0_stride lo
        g1[6] = 0;
        g1[7] = 0;
        v4i gz = (v4i){0, 0, 0, 0};
#if __clang_major__ >= 23
        v8i gz8 = (v8i){0, 0, 0, 0, 0, 0, 0, 0};
        __builtin_amdgcn_tensor_load_to_lds(g0, g1, gz, gz, gz8, 0);
#else
        __builtin_amdgcn_tensor_load_to_lds(g0, g1, gz, gz, 0);
#endif
      }
    }
  };

  auto stageA_sync = [&](int kkT) {   // fallback only
#pragma unroll 4
    for (int e = 0; e < 64; ++e) {
      int row = quad + 4 * e;
      float w = wgt[(long)row * KGEMM + kkT + sub];
      float q = (w > delta) ? wp : ((w < -delta) ? -wn : 0.0f);
      Alds[0][row * LSTRIDE + sub] = (_Float16)q;
    }
  };

  auto stageB = [&](int kkT) {
#pragma unroll
    for (int e = 0; e < 16; ++e) {
      int j  = quad + 4 * e;              // wave-uniform
      int kg = kkT + j;                   // uniform -> SALU
      int c  = kg / 9;
      int rs = kg - c * 9;
      int r  = rs / 3;
      int s  = rs - r * 3;
      int ih = hS + r - 1;
      int iw = wS + s - 1;
      float v = 0.0f;
      if ((unsigned)ih < (unsigned)W_IMG && (unsigned)iw < (unsigned)W_IMG)
        v = x[xBase + (long)c * HW_IMG + (r - 1) * W_IMG + (s - 1)];
      Blds[sub * LSTRIDE + j] = (_Float16)v;
    }
  };

  v8f acc[4][2];
#pragma unroll
  for (int mf = 0; mf < 4; ++mf)
#pragma unroll
    for (int nf = 0; nf < 2; ++nf) acc[mf][nf] = (v8f){};

  // ---- prologue: fill buffer 0 ----
  if constexpr (PRE) {
    tdmIssueA(0, 0);
    stageB(0);
    if (wv == 0) __builtin_amdgcn_s_wait_tensorcnt((short)0);
  } else {
    stageA_sync(0);
    stageB(0);
  }
  __syncthreads();

  int cur = 0;
  for (int kk = 0; kk < KGEMM; kk += KSTEP) {
    const bool hasNext = (kk + KSTEP) < KGEMM;
    if constexpr (PRE) {
      if (hasNext) tdmIssueA(kk + KSTEP, cur ^ 1);  // overlap with compute
    }

    // ---- compute: 2 k-subtiles x (4 M-frags x 2 N-frags) = 16 WMMAs ----
    const _Float16* Acur = &Alds[PRE ? cur : 0][0];
#pragma unroll
    for (int ks = 0; ks < KSTEP; ks += 32) {
      // B layout (16-bit table pattern): lane group g -> k = ks+8g..+7, ks+16+8g..+7
      v16h bfrag[2];
#pragma unroll
      for (int nf = 0; nf < 2; ++nf) {
        int bp = wnt * 32 + nf * 16 + ln;
        v8h blo = *(const v8h*)&Blds[bp * LSTRIDE + ks + 8 * g];
        v8h bhi = *(const v8h*)&Blds[bp * LSTRIDE + ks + 16 + 8 * g];
        bfrag[nf] = __builtin_shufflevector(blo, bhi,
            0,1,2,3,4,5,6,7,8,9,10,11,12,13,14,15);
      }
      // A 16-bit layout (ISA 7.12.2): lane group g -> k = ks+8g..+7, ks+16+8g..+7
#pragma unroll
      for (int mf = 0; mf < 4; ++mf) {
        int ar = wm * 64 + mf * 16 + ln;
        v8h alo = *(const v8h*)&Acur[ar * LSTRIDE + ks + 8 * g];
        v8h ahi = *(const v8h*)&Acur[ar * LSTRIDE + ks + 16 + 8 * g];
        v16h a = __builtin_shufflevector(alo, ahi,
            0,1,2,3,4,5,6,7,8,9,10,11,12,13,14,15);
        acc[mf][0] = __builtin_amdgcn_wmma_f32_16x16x32_f16(
            false, a, false, bfrag[0], (short)0, acc[mf][0], false, false);
        acc[mf][1] = __builtin_amdgcn_wmma_f32_16x16x32_f16(
            false, a, false, bfrag[1], (short)0, acc[mf][1], false, false);
      }
    }
    __syncthreads();                      // done reading Blds (and fallback A)

    if (hasNext) {
      if constexpr (!PRE) stageA_sync(kk + KSTEP);
      stageB(kk + KSTEP);
      if constexpr (PRE) {
        if (wv == 0) __builtin_amdgcn_s_wait_tensorcnt((short)0);
      }
      __syncthreads();                    // next B staged + next A tile landed
    }
    cur ^= 1;
  }

  // ---- epilogue: C/D layout m = i + 8g, n = ln; add bias, store NCHW ----
#pragma unroll
  for (int nf = 0; nf < 2; ++nf) {
    int pix = pixBase + wnt * 32 + nf * 16 + ln;
    int nb  = pix / HW_IMG;
    int hw  = pix - nb * HW_IMG;
    long obase = (long)nb * K_OUT * HW_IMG + hw;
#pragma unroll
    for (int mf = 0; mf < 4; ++mf) {
#pragma unroll
      for (int i = 0; i < 8; ++i) {
        int ch = wm * 64 + mf * 16 + i + 8 * g;
        out[obase + (long)ch * HW_IMG] = acc[mf][nf][i] + bias[ch];
      }
    }
  }
}

extern "C" void kernel_launch(void* const* d_in, const int* in_sizes, int n_in,
                              void* d_out, int out_size, void* d_ws, size_t ws_size,
                              hipStream_t stream) {
  const float* x    = (const float*)d_in[0];
  const float* wgt  = (const float*)d_in[1];
  const float* bias = (const float*)d_in[2];
  const float* Wp   = (const float*)d_in[3];
  const float* Wn   = (const float*)d_in[4];
  float* out = (float*)d_out;
  float* wsf = (float*)d_ws;                 // [0..511] partials, [512] max
  _Float16* wq = (_Float16*)((char*)d_ws + 4096);
  const size_t wsNeed = 4096 + sizeof(_Float16) * (size_t)NWEIGHT;

  ttq_absmax_stage1<<<512, 256, 0, stream>>>(wgt, NWEIGHT, wsf);
  ttq_absmax_stage2<<<1, 256, 0, stream>>>(wsf);

  dim3 grid(32 * HW_IMG / NTILE, 1);         // (1568, 1)
  if (ws_size >= wsNeed) {
    ttq_quantize<<<NWEIGHT / 1024, 256, 0, stream>>>(wgt, wsf + 512, Wp, Wn, wq);
    ttq_conv_wmma<true><<<grid, 256, 0, stream>>>(
        x, wgt, wq, bias, Wp, Wn, wsf + 512, out);
  } else {
    ttq_conv_wmma<false><<<grid, 256, 0, stream>>>(
        x, wgt, wq, bias, Wp, Wn, wsf + 512, out);
  }
}
